// EnhancedTransformerEncoder_66073776881985
// MI455X (gfx1250) — matrix operand order repaired and checked by reading, not compiled, not run on hardware
//
#include <hip/hip_runtime.h>
#include <hip/hip_bf16.h>
#include <cstdint>

// ---------------------------------------------------------------------------
// EnhancedTransformerEncoder for MI455X (gfx1250, wave32)
//   B=8, L=4096, D_MODEL=256, 8 heads x 32 dims, window 9 (+/-4)
// Pipeline:
//   0) convert W{q,k,v,o} and x fp32 -> f16          (elementwise)
//   1) Q/K/V = W @ x + b : WMMA f32_16x16x32_f16, fused over gridDim.z=3,
//      TDM-staged 256x64 B tiles in LDS, 4x4 register-tiled accumulators
//   2) RoPE + window-9 softmax attention (fp32 VALU, LDS tiles) -> f16
//   3) out = Wo @ attn + bo : same WMMA GEMM, fp32 output
// ---------------------------------------------------------------------------

#define D_MODEL 256
#define NH      8
#define HD      32
#define SEQ     4096
#define BATCH   8

typedef __attribute__((ext_vector_type(16))) _Float16     v16h;
typedef __attribute__((ext_vector_type(8)))  float        v8f;
typedef __attribute__((ext_vector_type(4)))  unsigned int v4u;
typedef __attribute__((ext_vector_type(8)))  int          v8i;
typedef __attribute__((ext_vector_type(4)))  int          v4i;

union F16Frag { uint4 u4[2]; uint2 u2[4]; v16h h; };

// 16 contiguous halves via four 8B loads (8B-aligned padded LDS rows)
__device__ __forceinline__ v16h ld16h_b64(const _Float16* p) {
  F16Frag t;
  const uint2* q = (const uint2*)p;
  t.u2[0] = q[0]; t.u2[1] = q[1]; t.u2[2] = q[2]; t.u2[3] = q[3];
  return t.h;
}
// A-fragment: two separate 8-half runs (K and K+16), each one 16B load
__device__ __forceinline__ v16h ld8h8h(const _Float16* p0, const _Float16* p1) {
  F16Frag t;
  t.u4[0] = *(const uint4*)p0;
  t.u4[1] = *(const uint4*)p1;
  return t.h;
}

// ---------------------------------------------------------------------------
// fp32 -> f16 convert
// ---------------------------------------------------------------------------
__global__ void cvt_f32_to_f16(const float* __restrict__ src,
                               _Float16* __restrict__ dst, int n) {
  int i = blockIdx.x * blockDim.x + threadIdx.x;
  if (i < n) dst[i] = (_Float16)src[i];
}

// ---------------------------------------------------------------------------
// Y[z][b][m][n] = sum_k W[z][m][k] * X[b][k][n] + bias_z[m]
//   W: z-concatenated [256][256] f16, X: [B][256][SEQ] f16, Y: f16 or f32
// block = 128 threads (4 waves); block tile = 256(M) x 64(N); K looped by 32.
// Each wave: 64(M) x 64(N) = 4x4 v8f accumulators; A-fragments reused across
// the 4 N-tiles in registers (4x less W traffic from L2 than a 16-wide tile).
// The 256x64 B tile is staged by the Tensor Data Mover with 2 dwords of LDS
// padding per 32-dword row -> 136B row stride (conflict-free lane reads).
// ---------------------------------------------------------------------------
#define XS_ROW 68   // halves per padded LDS row (64 data + 4 pad)
#define NTW    64   // N columns per block

__global__ __launch_bounds__(128) void gemm256_wmma(
    const _Float16* __restrict__ Wh0,   // z-concatenated weights
    const float* __restrict__ bias0,    // per-z bias pointers
    const float* __restrict__ bias1,
    const float* __restrict__ bias2,
    const _Float16* __restrict__ Xh,
    _Float16* __restrict__ outH0,       // z-concatenated f16 outputs
    float*    __restrict__ outF,        // fp32 output (writeF32 path)
    int writeF32)
{
  const int z    = blockIdx.z;
  const int b    = blockIdx.y;
  const int n0   = blockIdx.x * NTW;
  const int lane = threadIdx.x & 31;
  const int wave = threadIdx.x >> 5;
  const int m0w  = wave * 64;            // 4 waves x 64 rows = 256 rows

  const _Float16* __restrict__ Wh   = Wh0 + (size_t)z * D_MODEL * D_MODEL;
  const float*    __restrict__ bias = (z == 0) ? bias0 : ((z == 1) ? bias1 : bias2);

  __shared__ _Float16 Xs[D_MODEL * XS_ROW];   // 34 KB padded K-major tile

  const _Float16* __restrict__ Xb = Xh + (size_t)b * D_MODEL * SEQ;

#if defined(__gfx1250__)
  if (threadIdx.x < 32) {  // wave 0 issues one TDM descriptor for the tile
    const unsigned long long ga  = (unsigned long long)(uintptr_t)(Xb + n0);
    const unsigned int       lds = (unsigned int)(uintptr_t)(&Xs[0]);
    v4u g0;
    g0[0] = 1u;                                            // count=1, load
    g0[1] = lds;                                           // lds_addr
    g0[2] = (unsigned int)(ga & 0xffffffffull);            // global_addr lo
    g0[3] = (unsigned int)((ga >> 32) & 0x1ffffffull)      // global_addr hi
          | (2u << 30);                                    // type=2 (image)
    union { unsigned long long q[4]; v8i v; } g1;
    const unsigned long long N = SEQ, K = D_MODEL;
    g1.q[0] = (1ull << 16)                 // data_size = 2 bytes
            | (1ull << 20)                 // pad_enable
            | (4ull << 22)                 // pad_interval: every 32 dwords
            | (1ull << 25)                 // pad_amount: 2 dwords
            | ((N & 0xffffull) << 48);     // tensor_dim0[15:0]
    g1.q[1] = (N >> 16)                    // tensor_dim0[31:16]
            | (K << 16)                    // tensor_dim1 (bits 111:80)
            | ((unsigned long long)NTW << 48); // tile_dim0 (bits 127:112)
    g1.q[2] = K                            // tile_dim1   (bits 143:128)
            | ((N & 0xffffffffull) << 32); // tensor_dim0_stride lo
    g1.q[3] = (N >> 32);                   // tensor_dim0_stride hi
    v4i z4 = {0, 0, 0, 0};
#if __clang_major__ >= 23
    v8i z8 = {};
    __builtin_amdgcn_tensor_load_to_lds(g0, g1.v, z4, z4, z8, 0);
#else
    __builtin_amdgcn_tensor_load_to_lds(g0, g1.v, z4, z4, 0);
#endif
    __builtin_amdgcn_s_wait_tensorcnt(0);
    asm volatile("" ::: "memory");
  }
#else
  // host / non-gfx1250 pass: cooperative staging with the same padded layout
  for (int i = threadIdx.x; i < D_MODEL * 16; i += 128) {
    const int k = i >> 4, c = i & 15;
    ((uint2*)(Xs + (size_t)k * XS_ROW))[c] =
        ((const uint2*)(Xb + (size_t)k * SEQ + n0))[c];
  }
#endif
  __builtin_prefetch(Wh + (size_t)m0w * D_MODEL, 0, 0);  // global_prefetch_b8
  __syncthreads();

  v8f acc[4][4] = {};

  for (int k0 = 0; k0 < D_MODEL; k0 += 32) {
    // B fragments: lane&15 = N column; K = 8h+s (s<8), 16+8h+(s-8) (s>=8)
    const _Float16* brow = Xs + (size_t)(k0 + ((lane >> 4) << 3)) * XS_ROW + (lane & 15);
    v16h bf[4];
#pragma unroll
    for (int nt = 0; nt < 4; ++nt)
#pragma unroll
      for (int s = 0; s < 16; ++s)
        bf[nt][s] = brow[nt * 16 + (size_t)(s < 8 ? s : s + 8) * XS_ROW];

    const int mA   = (lane & 15);
    const int koff = k0 + ((lane >> 4) << 3);  // upper half-wave: K+8
#pragma unroll
    for (int mt = 0; mt < 4; ++mt) {
      const _Float16* wr = Wh + (size_t)(m0w + mt * 16 + mA) * D_MODEL;
      const v16h afrag = ld8h8h(wr + koff, wr + koff + 16);
#pragma unroll
      for (int nt = 0; nt < 4; ++nt)
        acc[mt][nt] = __builtin_amdgcn_wmma_f32_16x16x32_f16(
            false, afrag, false, bf[nt], (short)0, acc[mt][nt], false, false);
    }
  }

  // D layout: VGPR r -> M = r (+8 for lanes 16..31), lane&15 -> N
  const int nl     = (lane & 15);
  const int rowsel = (lane >> 4) << 3;
#pragma unroll
  for (int mt = 0; mt < 4; ++mt) {
#pragma unroll
    for (int nt = 0; nt < 4; ++nt) {
      const int n = n0 + nt * 16 + nl;
#pragma unroll
      for (int r = 0; r < 8; ++r) {
        const int m = m0w + mt * 16 + rowsel + r;
        const float val = acc[mt][nt][r] + bias[m];
        const size_t idx = ((size_t)b * D_MODEL + m) * SEQ + n;
        if (writeF32) outF[idx] = val;
        else          outH0[(size_t)z * BATCH * D_MODEL * SEQ + idx] = (_Float16)val;
      }
    }
  }
}

// ---------------------------------------------------------------------------
// RoPE + window-9 attention, fp32 math. One block = one (b, head, 128 l's).
// K/V halo tiles (RoPE applied to K) staged in LDS.
// ---------------------------------------------------------------------------
__global__ __launch_bounds__(128) void attn_window9(
    const _Float16* __restrict__ Qh,
    const _Float16* __restrict__ Kh,
    const _Float16* __restrict__ Vh,
    const unsigned char* __restrict__ mask,   // [B][SEQ] bool bytes
    _Float16* __restrict__ Oh)
{
  constexpr int TILE = 128, HALO = 4, ROWS = TILE + 2 * HALO;
  const int t    = threadIdx.x;
  const int b    = blockIdx.y >> 3;
  const int l0   = blockIdx.x * TILE;
  const size_t base = (size_t)blockIdx.y * HD * SEQ;   // (b*8+head)*32*L

  __shared__ float ks[ROWS][HD + 1];
  __shared__ float vs[ROWS][HD + 1];

  const float lg = 0.5756462732485114f;  // log(10000)/16

  for (int r = t; r < ROWS; r += 128) {
    const int  l   = l0 - HALO + r;
    const bool inb = (l >= 0) && (l < SEQ);
    const int  lc  = inb ? l : 0;
    float kr[HD], vr[HD];
#pragma unroll
    for (int dd = 0; dd < HD; ++dd) {
      kr[dd] = inb ? (float)Kh[base + (size_t)dd * SEQ + lc] : 0.f;
      vr[dd] = inb ? (float)Vh[base + (size_t)dd * SEQ + lc] : 0.f;
    }
    const float pos = (float)lc;
#pragma unroll
    for (int j = 0; j < HD / 2; ++j) {
      const float ang = pos * __expf(-lg * (float)j);
      float s, c; __sincosf(ang, &s, &c);
      const float x0 = kr[j], x1 = kr[j + HD / 2];
      kr[j]          = x0 * c - x1 * s;
      kr[j + HD / 2] = x1 * c + x0 * s;
    }
#pragma unroll
    for (int dd = 0; dd < HD; ++dd) { ks[r][dd] = kr[dd]; vs[r][dd] = vr[dd]; }
  }
  __syncthreads();

  const int l = l0 + t;
  float q[HD];
#pragma unroll
  for (int dd = 0; dd < HD; ++dd)
    q[dd] = (float)Qh[base + (size_t)dd * SEQ + l];
  {
    const float pos = (float)l;
#pragma unroll
    for (int j = 0; j < HD / 2; ++j) {
      const float ang = pos * __expf(-lg * (float)j);
      float s, c; __sincosf(ang, &s, &c);
      const float x0 = q[j], x1 = q[j + HD / 2];
      q[j]          = x0 * c - x1 * s;
      q[j + HD / 2] = x1 * c + x0 * s;
    }
  }

  const float scale = 0.17677669529663687f;  // 32^-0.5 (d^-0.25 applied twice)
  float sc[9];
#pragma unroll
  for (int j = 0; j < 9; ++j) {
    const int  kp    = l + j - HALO;
    const bool valid = (kp >= 0) && (kp < SEQ) &&
                       (mask[(size_t)b * SEQ + (kp < 0 ? 0 : (kp >= SEQ ? SEQ - 1 : kp))] != 0);
    float dot = 0.f;
#pragma unroll
    for (int dd = 0; dd < HD; ++dd) dot += q[dd] * ks[t + j][dd];
    sc[j] = valid ? dot * scale : -1.0e9f;
  }
  float mx = sc[0];
#pragma unroll
  for (int j = 1; j < 9; ++j) mx = fmaxf(mx, sc[j]);
  float p[9], sum = 0.f;
#pragma unroll
  for (int j = 0; j < 9; ++j) { p[j] = __expf(sc[j] - mx); sum += p[j]; }
  const float inv = 1.f / sum;
  const bool  qv  = mask[(size_t)b * SEQ + l] != 0;

#pragma unroll
  for (int dd = 0; dd < HD; ++dd) {
    float o = 0.f;
#pragma unroll
    for (int j = 0; j < 9; ++j) o += p[j] * vs[t + j][dd];
    o = qv ? o * inv : 0.f;
    Oh[base + (size_t)dd * SEQ + l] = (_Float16)o;
  }
}

// ---------------------------------------------------------------------------
extern "C" void kernel_launch(void* const* d_in, const int* in_sizes, int n_in,
                              void* d_out, int out_size, void* d_ws, size_t ws_size,
                              hipStream_t stream)
{
  const float*         x    = (const float*)d_in[0];
  const unsigned char* mask = (const unsigned char*)d_in[1];  // jnp bool -> 1B
  const float* Wq = (const float*)d_in[2];
  const float* bq = (const float*)d_in[3];
  const float* Wk = (const float*)d_in[4];
  const float* bk = (const float*)d_in[5];
  const float* Wv = (const float*)d_in[6];
  const float* bv = (const float*)d_in[7];
  const float* Wo = (const float*)d_in[8];
  const float* bo = (const float*)d_in[9];
  float* out = (float*)d_out;

  const size_t WSZ = (size_t)D_MODEL * D_MODEL;              // 65536
  const size_t TSZ = (size_t)BATCH * D_MODEL * SEQ;          // 8388608

  char* p = (char*)d_ws;
  _Float16* WqH = (_Float16*)p; p += WSZ * 2;   // Wq,Wk,Wv contiguous for z-index
  _Float16* WkH = (_Float16*)p; p += WSZ * 2;
  _Float16* WvH = (_Float16*)p; p += WSZ * 2;
  _Float16* WoH = (_Float16*)p; p += WSZ * 2;
  _Float16* Xh  = (_Float16*)p; p += TSZ * 2;
  _Float16* Qh  = (_Float16*)p; p += TSZ * 2;   // Q,K,V contiguous for z-index
  _Float16* Kh  = (_Float16*)p; p += TSZ * 2;
  _Float16* Vh  = (_Float16*)p; p += TSZ * 2;
  _Float16* Ah  = (_Float16*)p; p += TSZ * 2;   // ~84.4 MB total

  cvt_f32_to_f16<<<(int)(WSZ / 256), 256, 0, stream>>>(Wq, WqH, (int)WSZ);
  cvt_f32_to_f16<<<(int)(WSZ / 256), 256, 0, stream>>>(Wk, WkH, (int)WSZ);
  cvt_f32_to_f16<<<(int)(WSZ / 256), 256, 0, stream>>>(Wv, WvH, (int)WSZ);
  cvt_f32_to_f16<<<(int)(WSZ / 256), 256, 0, stream>>>(Wo, WoH, (int)WSZ);
  cvt_f32_to_f16<<<(int)(TSZ / 256), 256, 0, stream>>>(x,  Xh,  (int)TSZ);

  // Fused Q/K/V projection: gridDim.z selects weight/bias/output
  dim3 gqkv(SEQ / NTW, BATCH, 3);
  gemm256_wmma<<<gqkv, 128, 0, stream>>>(WqH, bq, bk, bv, Xh, Qh, nullptr, 0);

  dim3 ga(SEQ / 128, BATCH * NH);
  attn_window9<<<ga, 128, 0, stream>>>(Qh, Kh, Vh, mask, Ah);

  // Output projection, fp32 result
  dim3 go(SEQ / NTW, BATCH, 1);
  gemm256_wmma<<<go, 128, 0, stream>>>(WoH, bo, bo, bo, Ah, nullptr, out, 1);
}
